// GraphMemoryPoolV2_78202764525974
// MI455X (gfx1250) — matrix operand order repaired and checked by reading, not compiled, not run on hardware
//
#include <hip/hip_runtime.h>
#include <hip/hip_bf16.h>

typedef __attribute__((ext_vector_type(16))) __bf16 v16bf;
typedef __attribute__((ext_vector_type(8)))  float  v8f;
typedef int v4i_ __attribute__((vector_size(16)));
typedef __bf16 bf16;

#define BN_TOT 32768            // B*N = 32*1024
#define NN     1024
#define MM     16
#define DD     768
#define D3     2304

// ---------------------------------------------------------------------------
// CDNA5 async global->LDS copy (ASYNCcnt-tracked), with safe fallback
// ---------------------------------------------------------------------------
#if defined(__HIP_DEVICE_COMPILE__) && defined(__has_builtin)
#  if __has_builtin(__builtin_amdgcn_global_load_async_to_lds_b128)
#    define HAVE_ASYNC_LDS 1
#  else
#    define HAVE_ASYNC_LDS 0
#  endif
#else
#  define HAVE_ASYNC_LDS 0
#endif

__device__ __forceinline__ void cp_async_b128(const bf16* g, bf16* l)
{
#if HAVE_ASYNC_LDS
    __builtin_amdgcn_global_load_async_to_lds_b128(
        (__attribute__((address_space(1))) v4i_*)(v4i_*)(void*)const_cast<bf16*>(g),
        (__attribute__((address_space(3))) v4i_*)(v4i_*)(void*)l,
        0, 0);
#else
    *(uint4*)l = *(const uint4*)g;
#endif
}

__device__ __forceinline__ void wait_async_lds()
{
#if HAVE_ASYNC_LDS
#  if __has_builtin(__builtin_amdgcn_s_wait_asynccnt)
    __builtin_amdgcn_s_wait_asynccnt(0);
#  else
    asm volatile("s_wait_asynccnt 0x0" ::: "memory");
#  endif
#endif
}

// ---------------------------------------------------------------------------
// elementwise f32 -> bf16 cast with independent src/dst row strides
// ---------------------------------------------------------------------------
__global__ __launch_bounds__(256) void cast_bf16_strided(
    const float* __restrict__ src, bf16* __restrict__ dst,
    long rows, long cols, long sld, long dld)
{
    long i = (long)blockIdx.x * 256 + threadIdx.x;
    long total = rows * cols;
    if (i >= total) return;
    long r = i / cols, c = i - r * cols;
    dst[r * dld + c] = (bf16)src[r * sld + c];
}

// ---------------------------------------------------------------------------
// adjacency normalization: d = (rowsum(adj + I))^-1/2 ; A' = d_i (a+I) d_j
// ---------------------------------------------------------------------------
__global__ __launch_bounds__(256) void adj_rowsum(
    const float* __restrict__ adj, float* __restrict__ dinv, int n)
{
    int row = blockIdx.x;
    float s = 0.f;
    for (int j = threadIdx.x; j < n; j += 256)
        s += adj[(long)row * n + j] + (j == row ? 1.f : 0.f);
    __shared__ float red[256];
    red[threadIdx.x] = s;
    __syncthreads();
    for (int st = 128; st > 0; st >>= 1) {
        if (threadIdx.x < st) red[threadIdx.x] += red[threadIdx.x + st];
        __syncthreads();
    }
    if (threadIdx.x == 0) {
        float rs = red[0];
        dinv[row] = (rs > 0.f) ? rsqrtf(rs) : 0.f;
    }
}

__global__ __launch_bounds__(256) void adj_scale(
    const float* __restrict__ adj, const float* __restrict__ dinv,
    bf16* __restrict__ outb, int n)
{
    long i = (long)blockIdx.x * 256 + threadIdx.x;
    if (i >= (long)n * n) return;
    int r = (int)(i / n), c = (int)(i - (long)r * n);
    float a = adj[i] + (r == c ? 1.f : 0.f);
    outb[i] = (bf16)(dinv[r] * a * dinv[c]);
}

// ---------------------------------------------------------------------------
// WMMA bf16 GEMM: C[M,N] = A[M,K] * B + bias, f32 accumulate.
// BT=1: B stored [N,K] row-major (weights, used as B^T)  -> async LDS staging
// BT=0: B stored [K,N] row-major (transpose-scattered into LDS)
// EPI: 0 = store, 1 = +bias, 2 = +bias then exact gelu
// Block tile 64(M) x 128(N), BK=32, double-buffered LDS.
// 256 threads = 8 waves, each wave computes a 32x32 tile (4 WMMAs / k-step).
// ---------------------------------------------------------------------------
__device__ __forceinline__ float gelu_exact(float v)
{
    return 0.5f * v * (1.0f + erff(v * 0.70710678118654752f));
}

template<int BT, int EPI>
__global__ __launch_bounds__(256) void gemm_wmma_bf16(
    const bf16* __restrict__ A, long lda,
    const bf16* __restrict__ B, long ldb,
    const float* __restrict__ bias,
    float* __restrict__ C, long ldc,
    int K)
{
    __shared__ alignas(16) bf16 lA[2][64][32];
    __shared__ alignas(16) bf16 lB[2][128][32];   // always [n][k], K contiguous

    const int tid  = threadIdx.x;
    const int lane = tid & 31;
    const int wave = tid >> 5;       // 0..7
    const int wm   = wave & 1;       // row half of block tile
    const int wn   = wave >> 1;      // 0..3 column quarter
    const long bm  = (long)blockIdx.x * 64;
    const long bn  = (long)blockIdx.y * 128;

    v8f acc[2][2] = {};

    auto stage = [&](int buf, int k0) {
        // A tile: 64x32 bf16 (4 KB), one async 16B transfer per thread
        {
            int r = tid >> 2, c = (tid & 3) * 8;
            cp_async_b128(A + (bm + r) * lda + k0 + c, &lA[buf][r][c]);
        }
        if (BT) {
            int r = tid >> 2, c = (tid & 3) * 8;
            cp_async_b128(B + (bn + r) * ldb + k0 + c,      &lB[buf][r][c]);
            cp_async_b128(B + (bn + r + 64) * ldb + k0 + c, &lB[buf][r + 64][c]);
        } else {
            // [K,N] global -> [n][k] LDS transpose scatter (VGPR-staged)
            int kr = tid >> 4, nb = (tid & 15) * 8;
            bf16 t0[8], t1[8];
            *(uint4*)t0 = *(const uint4*)(B + (long)(k0 + kr)      * ldb + bn + nb);
            *(uint4*)t1 = *(const uint4*)(B + (long)(k0 + kr + 16) * ldb + bn + nb);
#pragma unroll
            for (int i = 0; i < 8; ++i) {
                lB[buf][nb + i][kr]      = t0[i];
                lB[buf][nb + i][kr + 16] = t1[i];
            }
        }
    };

    const int nsteps = K >> 5;

    stage(0, 0);
    wait_async_lds();
    __syncthreads();

    for (int s = 0; s < nsteps; ++s) {
        const int buf = s & 1;
        const int k0  = s << 5;
        if (s + 1 < nsteps) {
            stage(buf ^ 1, k0 + 32);                 // overlap with compute below
            __builtin_prefetch(A + (bm + (tid >> 2)) * lda + k0 + 64, 0, 1);
        }

        // ---- fragments + 4 WMMAs per wave ----
        // A 16x32 bf16 layout: lanes 0-15 row M=lane hold K 0-7 & 16-23;
        //                      lanes 16-31 row M=lane-16 hold K 8-15 & 24-31.
        const int arow0 = wm * 32 + (lane & 15);
        const int akl   = (lane < 16) ? 0 : 8;
        // B 32x16 bf16 layout: lanes 0-15 col N=lane K 0-7 & 16-23; lanes 16-31 K 8-15 & 24-31.
        const int bcol0 = wn * 32 + (lane & 15);
        const int bkl   = (lane < 16) ? 0 : 8;

        v16bf afr[2], bfr[2];
#pragma unroll
        for (int i = 0; i < 2; ++i) {
            int row = arow0 + i * 16;
            ((uint4*)&afr[i])[0] = *(const uint4*)&lA[buf][row][akl];
            ((uint4*)&afr[i])[1] = *(const uint4*)&lA[buf][row][akl + 16];
        }
#pragma unroll
        for (int j = 0; j < 2; ++j) {
            int col = bcol0 + j * 16;
            ((uint4*)&bfr[j])[0] = *(const uint4*)&lB[buf][col][bkl];
            ((uint4*)&bfr[j])[1] = *(const uint4*)&lB[buf][col][bkl + 16];
        }
#pragma unroll
        for (int i = 0; i < 2; ++i)
#pragma unroll
            for (int j = 0; j < 2; ++j)
                acc[i][j] = __builtin_amdgcn_wmma_f32_16x16x32_bf16(
                    false, afr[i], false, bfr[j],
                    (short)0, acc[i][j], false, false);

        if (s + 1 < nsteps) wait_async_lds();        // own async copies done
        __syncthreads();                             // everyone's copies visible
    }

    // ---- epilogue: C 16x16 f32 layout: VGPR r, lanes0-15 M=r, lanes16-31 M=r+8
#pragma unroll
    for (int i = 0; i < 2; ++i) {
#pragma unroll
        for (int j = 0; j < 2; ++j) {
            long cn    = bn + wn * 32 + j * 16 + (lane & 15);
            long rbase = bm + wm * 32 + i * 16 + ((lane >> 4) * 8);
            float bv = (EPI > 0) ? bias[cn] : 0.f;
#pragma unroll
            for (int r = 0; r < 8; ++r) {
                float v = acc[i][j][r] + bv;
                if (EPI == 2) v = gelu_exact(v);
                C[(rbase + r) * ldc + cn] = v;
            }
        }
    }
}

// ---------------------------------------------------------------------------
// top-R memory read: per (b,n) — cosine sims over M=16 keys, top-4 softmax
// (tau=0.7), weighted val read. Writes f32 and bf16 (into cat, stride 2304).
// One 128-thread block per (b,n) row.
// ---------------------------------------------------------------------------
__global__ __launch_bounds__(128) void topr_read(
    const float* __restrict__ q,
    const float* __restrict__ keys,   // [N, M, D]
    const float* __restrict__ vals,   // [N, M, D]
    float* __restrict__ outf,
    bf16*  __restrict__ outb, long outb_ld)
{
    const int bn = blockIdx.x;
    const int n  = bn & (NN - 1);
    const float* qr = q    + (long)bn * DD;
    const float* kb = keys + (long)n * MM * DD;
    const float* vb = vals + (long)n * MM * DD;

    float qsq = 0.f, dot[MM] = {}, ksq[MM] = {};
    for (int d = threadIdx.x; d < DD; d += 128) {
        float qv = qr[d];
        qsq += qv * qv;
#pragma unroll
        for (int m = 0; m < MM; ++m) {
            float kv = kb[m * DD + d];
            dot[m] += qv * kv;
            ksq[m] += kv * kv;
        }
    }

    __shared__ float part[33][128];
    part[0][threadIdx.x] = qsq;
#pragma unroll
    for (int m = 0; m < MM; ++m) {
        part[1 + m][threadIdx.x]  = dot[m];
        part[17 + m][threadIdx.x] = ksq[m];
    }
    __syncthreads();
    if (threadIdx.x < 33) {
        float s = 0.f;
        for (int i = 0; i < 128; ++i) s += part[threadIdx.x][i];
        part[threadIdx.x][0] = s;
    }
    __syncthreads();

    __shared__ float alpha[4];
    __shared__ int   aidx[4];
    if (threadIdx.x == 0) {
        float qn = fmaxf(sqrtf(part[0][0]), 1e-12f);
        float sim[MM];
#pragma unroll
        for (int m = 0; m < MM; ++m) {
            float kn = fmaxf(sqrtf(part[17 + m][0]), 1e-12f);
            sim[m] = part[1 + m][0] / (qn * kn);
        }
        bool used[MM] = {};
        float tv[4]; int ti[4];
        for (int r = 0; r < 4; ++r) {
            float best = -3.4e38f; int bi = 0;
            for (int m = 0; m < MM; ++m)
                if (!used[m] && sim[m] > best) { best = sim[m]; bi = m; }
            used[bi] = true; tv[r] = best; ti[r] = bi;
        }
        float mx = tv[0], e[4], se = 0.f;
        for (int r = 0; r < 4; ++r) { e[r] = __expf((tv[r] - mx) / 0.7f); se += e[r]; }
        for (int r = 0; r < 4; ++r) { alpha[r] = e[r] / se; aidx[r] = ti[r]; }
    }
    __syncthreads();

    float a0 = alpha[0], a1 = alpha[1], a2 = alpha[2], a3 = alpha[3];
    long  i0 = aidx[0],  i1 = aidx[1],  i2 = aidx[2],  i3 = aidx[3];
    for (int d = threadIdx.x; d < DD; d += 128) {
        float v = a0 * vb[i0 * DD + d] + a1 * vb[i1 * DD + d]
                + a2 * vb[i2 * DD + d] + a3 * vb[i3 * DD + d];
        outf[(long)bn * DD + d] = v;
        outb[(long)bn * outb_ld + d] = (bf16)v;
    }
}

// ---------------------------------------------------------------------------
// fusion: per row, logits = h . f2_w^T + f2_b (3 dots), softmax,
// fused = w0*x + w1*self_mem + w2*nbr_mem  (stored bf16 for the out-GEMM)
// ---------------------------------------------------------------------------
__global__ __launch_bounds__(128) void fuse_rows(
    const float* __restrict__ h,
    const float* __restrict__ f2w, const float* __restrict__ f2b,
    const float* __restrict__ x,
    const float* __restrict__ sm, const float* __restrict__ nm,
    bf16* __restrict__ fused)
{
    const int bn = blockIdx.x;
    const float* hr = h + (long)bn * DD;
    float d0 = 0.f, d1 = 0.f, d2 = 0.f;
    for (int d = threadIdx.x; d < DD; d += 128) {
        float hv = hr[d];
        d0 += hv * f2w[d];
        d1 += hv * f2w[DD + d];
        d2 += hv * f2w[2 * DD + d];
    }
    __shared__ float red[3][128];
    red[0][threadIdx.x] = d0; red[1][threadIdx.x] = d1; red[2][threadIdx.x] = d2;
    __syncthreads();
    __shared__ float w[3];
    if (threadIdx.x < 3) {
        float s = 0.f;
        for (int i = 0; i < 128; ++i) s += red[threadIdx.x][i];
        red[threadIdx.x][0] = s + f2b[threadIdx.x];
    }
    __syncthreads();
    if (threadIdx.x == 0) {
        float l0 = red[0][0], l1 = red[1][0], l2 = red[2][0];
        float mx = fmaxf(l0, fmaxf(l1, l2));
        float e0 = __expf(l0 - mx), e1 = __expf(l1 - mx), e2 = __expf(l2 - mx);
        float se = e0 + e1 + e2;
        w[0] = e0 / se; w[1] = e1 / se; w[2] = e2 / se;
    }
    __syncthreads();
    float w0 = w[0], w1 = w[1], w2 = w[2];
    long base = (long)bn * DD;
    for (int d = threadIdx.x; d < DD; d += 128)
        fused[base + d] = (bf16)(w0 * x[base + d] + w1 * sm[base + d] + w2 * nm[base + d]);
}

// ---------------------------------------------------------------------------
// final: y = LayerNorm(x + out)   (out already gelu'd by GEMM epilogue)
// ---------------------------------------------------------------------------
__global__ __launch_bounds__(128) void ln_residual(
    const float* __restrict__ x, const float* __restrict__ o,
    const float* __restrict__ g, const float* __restrict__ b,
    float* __restrict__ y)
{
    const int bn = blockIdx.x;
    long base = (long)bn * DD;
    float s = 0.f, s2 = 0.f;
    for (int d = threadIdx.x; d < DD; d += 128) {
        float v = x[base + d] + o[base + d];
        s += v; s2 += v * v;
    }
    __shared__ float r1[128], r2[128];
    r1[threadIdx.x] = s; r2[threadIdx.x] = s2;
    __syncthreads();
    for (int st = 64; st > 0; st >>= 1) {
        if (threadIdx.x < st) {
            r1[threadIdx.x] += r1[threadIdx.x + st];
            r2[threadIdx.x] += r2[threadIdx.x + st];
        }
        __syncthreads();
    }
    float mu  = r1[0] * (1.0f / DD);
    float var = r2[0] * (1.0f / DD) - mu * mu;
    float rstd = rsqrtf(var + 1e-5f);
    for (int d = threadIdx.x; d < DD; d += 128) {
        float v = x[base + d] + o[base + d];
        y[base + d] = (v - mu) * rstd * g[d] + b[d];
    }
}

// ---------------------------------------------------------------------------
extern "C" void kernel_launch(void* const* d_in, const int* in_sizes, int n_in,
                              void* d_out, int out_size, void* d_ws, size_t ws_size,
                              hipStream_t stream)
{
    const float* x        = (const float*)d_in[0];
    const float* adj      = (const float*)d_in[1];
    const float* mem_keys = (const float*)d_in[2];
    const float* mem_vals = (const float*)d_in[3];
    const float* q_w      = (const float*)d_in[4];
    const float* q_b      = (const float*)d_in[5];
    const float* f1_w     = (const float*)d_in[6];
    const float* f1_b     = (const float*)d_in[7];
    const float* f2_w     = (const float*)d_in[8];
    const float* f2_b     = (const float*)d_in[9];
    const float* out_w    = (const float*)d_in[10];
    const float* out_b    = (const float*)d_in[11];
    const float* ln_g     = (const float*)d_in[12];
    const float* ln_b     = (const float*)d_in[13];
    float* y = (float*)d_out;

    const long NMD = (long)NN * MM * DD;

    char* ws = (char*)d_ws;
    size_t off = 0;
    auto take = [&](size_t bytes) -> char* {
        char* p = ws + off;
        off += (bytes + 255) & ~(size_t)255;
        return p;
    };

    bf16*  cat    = (bf16*)take((size_t)BN_TOT * D3 * 2);     // [BN, 3D] bf16
    float* qbuf   = (float*)take((size_t)BN_TOT * DD * 4);    // q ; later h
    bf16*  qw16   = (bf16*)take((size_t)DD * DD * 2);
    bf16*  f1w16  = (bf16*)take((size_t)DD * D3 * 2);
    bf16*  outw16 = (bf16*)take((size_t)DD * DD * 2);
    bf16*  keys16 = (bf16*)take((size_t)NMD * 2);             // later fused (with vals16)
    bf16*  vals16 = (bf16*)take((size_t)NMD * 2);
    bf16*  adjn   = (bf16*)take((size_t)NN * NN * 2);
    float* dinv   = (float*)take((size_t)NN * 4);
    float* nbrk   = (float*)take((size_t)NMD * 4);            // later out_pre (with nbrv)
    float* nbrv   = (float*)take((size_t)NMD * 4);
    float* selfm  = (float*)take((size_t)BN_TOT * DD * 4);
    float* nbrm   = (float*)take((size_t)BN_TOT * DD * 4);
    (void)vals16; (void)nbrv;

    float* hbuf   = qbuf;      // alias: q dead after both topr passes
    bf16*  fused  = keys16;    // alias: keys16+vals16 region (contiguous, 48 MB)
    float* outpre = nbrk;      // alias: nbrk+nbrv region (contiguous, 96 MB)

    auto gblk = [](long n) { return (unsigned)((n + 255) / 256); };

    // 1) casts into bf16 staging
    cast_bf16_strided<<<gblk((long)BN_TOT * DD), 256, 0, stream>>>(x, cat, BN_TOT, DD, DD, D3);
    cast_bf16_strided<<<gblk((long)DD * DD), 256, 0, stream>>>(q_w, qw16, DD, DD, DD, DD);
    cast_bf16_strided<<<gblk((long)DD * D3), 256, 0, stream>>>(f1_w, f1w16, DD, D3, D3, D3);
    cast_bf16_strided<<<gblk((long)DD * DD), 256, 0, stream>>>(out_w, outw16, DD, DD, DD, DD);
    cast_bf16_strided<<<gblk(NMD), 256, 0, stream>>>(mem_keys, keys16, (long)NN * MM, DD, DD, DD);
    cast_bf16_strided<<<gblk(NMD), 256, 0, stream>>>(mem_vals, vals16, (long)NN * MM, DD, DD, DD);

    // 2) adjacency normalization
    adj_rowsum<<<NN, 256, 0, stream>>>(adj, dinv, NN);
    adj_scale<<<gblk((long)NN * NN), 256, 0, stream>>>(adj, dinv, adjn, NN);

    // 3) q = x @ q_w^T + q_b          (32768 x 768 x 768)
    gemm_wmma_bf16<1, 1><<<dim3(BN_TOT / 64, DD / 128), 256, 0, stream>>>(
        cat, D3, qw16, DD, q_b, qbuf, DD, DD);

    // 4) nbr_keys / nbr_vals = adj_norm @ mem_{keys,vals}   (1024 x 12288 x 1024)
    gemm_wmma_bf16<0, 0><<<dim3(NN / 64, (MM * DD) / 128), 256, 0, stream>>>(
        adjn, NN, keys16, (long)MM * DD, nullptr, nbrk, (long)MM * DD, NN);
    gemm_wmma_bf16<0, 0><<<dim3(NN / 64, (MM * DD) / 128), 256, 0, stream>>>(
        adjn, NN, vals16, (long)MM * DD, nullptr, nbrv, (long)MM * DD, NN);

    // 5) top-4 routed memory reads (self and neighbor)
    topr_read<<<BN_TOT, 128, 0, stream>>>(qbuf, mem_keys, mem_vals, selfm, cat + DD, D3);
    topr_read<<<BN_TOT, 128, 0, stream>>>(qbuf, nbrk, nbrv, nbrm, cat + 2 * DD, D3);

    // 6) h = gelu(cat @ f1_w^T + f1_b)     (32768 x 768 x 2304)  [h aliases q]
    gemm_wmma_bf16<1, 2><<<dim3(BN_TOT / 64, DD / 128), 256, 0, stream>>>(
        cat, D3, f1w16, D3, f1_b, hbuf, DD, D3);

    // 7) fusion softmax + weighted mix -> fused (bf16)
    fuse_rows<<<BN_TOT, 128, 0, stream>>>(hbuf, f2_w, f2_b, x, selfm, nbrm, fused);

    // 8) out = gelu(fused @ out_w^T + out_b)   (32768 x 768 x 768)
    gemm_wmma_bf16<1, 2><<<dim3(BN_TOT / 64, DD / 128), 256, 0, stream>>>(
        fused, DD, outw16, DD, out_b, outpre, DD, DD);

    // 9) y = LayerNorm(x + out)
    ln_residual<<<BN_TOT, 128, 0, stream>>>(x, outpre, ln_g, ln_b, y);

    (void)in_sizes; (void)n_in; (void)out_size; (void)ws_size;
}
